// NaturalSpeechGeneratorLoss_35321811042342
// MI455X (gfx1250) — matrix operand order repaired and checked by reading, not compiled, not run on hardware
//
#include <hip/hip_runtime.h>

// ---------------------------------------------------------------------------
// NaturalSpeech generator loss on gfx1250 (MI455X), wave32 + WMMA.
// ---------------------------------------------------------------------------

typedef _Float16 v16h __attribute__((ext_vector_type(16)));
typedef _Float16 v8h  __attribute__((ext_vector_type(8)));
typedef float    v8f  __attribute__((ext_vector_type(8)));

#define GAMMA_F   0.01f
#define WARP_F    134.4f
#define BIG_F     1.0e9f
#define MASKINF_F 100000.0f

// scr slots: 0 gen, 1 gen_e2e, 2 fm, 3 mel, 4 dtw_dir0, 5 dtw_dir1, 6 pmask_sum, 7 zmask_sum
__global__ void zero_scr_kernel(float* scr) {
    if (threadIdx.x < 16) scr[threadIdx.x] = 0.0f;
}

// mode 0: sum (1-x)^2 ; mode 1: sum |x-y|
__global__ void reduce_sum_kernel(const float* __restrict__ x, const float* __restrict__ y,
                                  long n, int mode, float* acc) {
    __shared__ float sm[256];
    float s = 0.0f;
    long stride = (long)gridDim.x * blockDim.x;
    for (long i = (long)blockIdx.x * blockDim.x + threadIdx.x; i < n; i += stride) {
        if (mode == 0) { float d = 1.0f - x[i]; s += d * d; }
        else           { s += fabsf(x[i] - y[i]); }
    }
    sm[threadIdx.x] = s;
    __syncthreads();
    for (int st = 128; st > 0; st >>= 1) {
        if (threadIdx.x < st) sm[threadIdx.x] += sm[threadIdx.x + st];
        __syncthreads();
    }
    if (threadIdx.x == 0) atomicAdd(acc, sm[0]);
}

__global__ void mask_sum_kernel(const int* __restrict__ x, long n, float* acc) {
    __shared__ float sm[256];
    float s = 0.0f;
    long stride = (long)gridDim.x * blockDim.x;
    for (long i = (long)blockIdx.x * blockDim.x + threadIdx.x; i < n; i += stride)
        s += (float)x[i];
    sm[threadIdx.x] = s;
    __syncthreads();
    for (int st = 128; st > 0; st >>= 1) {
        if (threadIdx.x < st) sm[threadIdx.x] += sm[threadIdx.x + st];
        __syncthreads();
    }
    if (threadIdx.x == 0) atomicAdd(acc, sm[0]);
}

// Pack per-row features for the fused KL GEMM.
//  A'[b,p,k] : k<H -> a=exp(-2 lp) ; k>=H -> -2*a*m          (f16)
//  B'[b,q,k] : k<H -> z^2          ; k>=H -> z               (f16)
//  rowterm[b,p] = sum_h lp + 0.5*sum_h a*m^2
//  colterm[b,q] = -sum_h lq
__global__ void pack_kernel(const float* __restrict__ z, const float* __restrict__ lq,
                            const float* __restrict__ m, const float* __restrict__ lp,
                            unsigned short* Apack_, unsigned short* Bpack_,
                            float* rowterm, float* colterm,
                            int B_, int H_, int T_, int P16) {
    int idx = blockIdx.x * blockDim.x + threadIdx.x;
    int total = B_ * P16;
    if (idx >= total) return;
    int b = idx / P16, p = idx % P16;
    int K2 = 2 * H_;
    _Float16* arow = reinterpret_cast<_Float16*>(Apack_) + (size_t)(b * P16 + p) * K2;
    _Float16* brow = reinterpret_cast<_Float16*>(Bpack_) + (size_t)(b * P16 + p) * K2;
    if (p >= T_) {  // zero-pad tile rows
        for (int k = 0; k < K2; ++k) { arow[k] = (_Float16)0.0f; brow[k] = (_Float16)0.0f; }
        return;
    }
    const float* lpb = lp + (size_t)b * H_ * T_ + p;
    const float* mb  = m  + (size_t)b * H_ * T_ + p;
    const float* zb  = z  + (size_t)b * H_ * T_ + p;
    const float* lqb = lq + (size_t)b * H_ * T_ + p;
    float rsum = 0.0f, lpsum = 0.0f, lqsum = 0.0f;
    for (int h = 0; h < H_; ++h) {
        float lpv = lpb[(size_t)h * T_];
        float mv  = mb [(size_t)h * T_];
        float a   = __expf(-2.0f * lpv);
        arow[h]       = (_Float16)a;
        arow[H_ + h]  = (_Float16)(-2.0f * a * mv);
        rsum += a * mv * mv;
        lpsum += lpv;
        float zv  = zb [(size_t)h * T_];
        float lqv = lqb[(size_t)h * T_];
        brow[h]      = (_Float16)(zv * zv);
        brow[H_ + h] = (_Float16)zv;
        lqsum += lqv;
    }
    rowterm[b * P16 + p] = lpsum + 0.5f * rsum;
    colterm[b * P16 + p] = -lqsum;
}

// One wave per 16x16 output tile. kl[b,p,q] = rowterm + colterm - 0.5H + 0.5*(A'.B'^T), masked.
__global__ void kl_gemm_wmma_kernel(const unsigned short* Apack_, const unsigned short* Bpack_,
                                    const float* __restrict__ rowterm, const float* __restrict__ colterm,
                                    const int* __restrict__ pmask, const int* __restrict__ qmask,
                                    float* __restrict__ kl,
                                    int B_, int H_, int T_, int P16) {
    const _Float16* Apack = reinterpret_cast<const _Float16*>(Apack_);
    const _Float16* Bpack = reinterpret_cast<const _Float16*>(Bpack_);
    const int K2 = 2 * H_;
    const int nt = P16 / 16;
    const int tilesPerB = nt * nt;
    const int totalTiles = B_ * tilesPerB;
    int wid = blockIdx.x * 8 + (threadIdx.x >> 5);
    if (wid >= totalTiles) return;            // wave-uniform exit (EXEC all-1 for WMMA)
    int b = wid / tilesPerB;
    int rem = wid % tilesPerB;
    int tm = rem / nt, tn = rem % nt;
    int lane = threadIdx.x & 31;
    int hlf = lane >> 4, lm = lane & 15;

    // A: 16x32 f16 tile. lane<16: M=lm, K chunks [0..7] & [16..23]; lane>=16: chunks [8..15] & [24..31]
    const _Float16* arow = Apack + (size_t)(b * P16 + tm * 16 + lm) * K2 + hlf * 8;
    // B: 32x16 f16 tile, stored [q,k]. lane<16: N=lm, K chunks [0..7] & [16..23]; lane>=16: chunks [8..15] & [24..31]
    const _Float16* brow = Bpack + (size_t)(b * P16 + tn * 16 + lm) * K2 + hlf * 8;

    v8f acc = {};
    for (int k0 = 0; k0 < K2; k0 += 32) {
        v8h a0 = *(const v8h*)(arow + k0);
        v8h a1 = *(const v8h*)(arow + k0 + 16);
        v8h b0 = *(const v8h*)(brow + k0);
        v8h b1 = *(const v8h*)(brow + k0 + 16);
        v16h av, bv;
#pragma unroll
        for (int i = 0; i < 8; ++i) {
            av[i] = a0[i]; av[8 + i] = a1[i];
            bv[i] = b0[i]; bv[8 + i] = b1[i];
        }
        acc = __builtin_amdgcn_wmma_f32_16x16x32_f16(false, av, false, bv,
                                                     (short)0, acc, false, false);
    }

    const int Tp1 = T_ + 1;
    int q = tn * 16 + lm;
    if (q >= T_) return;
    int qbv = (qmask[b * T_ + q] != 0);
    float ct = colterm[b * P16 + q];
    float hterm = -0.5f * (float)H_;
#pragma unroll
    for (int v = 0; v < 8; ++v) {
        int p = tm * 16 + hlf * 8 + v;   // C/D layout: M = v + 8*half, N = lane&15
        if (p < T_) {
            float val = 0.5f * acc[v] + rowterm[b * P16 + p] + ct + hterm;
            int pbv = (pmask[b * T_ + p] != 0);
            if (pbv ^ qbv)            val = MASKINF_F;
            else if (!pbv && !qbv)    val = 0.0f;
            kl[(size_t)b * Tp1 * Tp1 + (size_t)p * Tp1 + q] = val;
        }
    }
}

// Fill padded row T and column T (pad value 0, then mask with pb/qb = 0 at index T).
__global__ void pad_kernel(float* kl, const int* __restrict__ pmask, const int* __restrict__ qmask,
                           int B_, int T_) {
    int Tp1 = T_ + 1;
    int idx = blockIdx.x * blockDim.x + threadIdx.x;
    int total = B_ * Tp1;
    if (idx >= total) return;
    int b = idx / Tp1, t = idx % Tp1;
    float vr = (t < T_ && qmask[b * T_ + t]) ? MASKINF_F : 0.0f;  // p == T row
    kl[(size_t)b * Tp1 * Tp1 + (size_t)T_ * Tp1 + t] = vr;
    float vc = (t < T_ && pmask[b * T_ + t]) ? MASKINF_F : 0.0f;  // q == T col
    kl[(size_t)b * Tp1 * Tp1 + (size_t)t * Tp1 + T_] = vc;
}

// Soft-DTW wavefront: one block per (dir, batch). 3 rotating LDS diagonals.
__global__ void dtw_kernel(const float* __restrict__ kl, int B_, int T_, size_t klStride,
                           float* scr) {
    const int n = T_ + 1;           // n == m == T+1 (<= 512 assumed)
    const int Kd = 2 * n - 1;
    __shared__ float buf[3][512];
    int tid = threadIdx.x;
    int dir = blockIdx.x / B_;
    int b   = blockIdx.x % B_;
    const float* D = kl + (size_t)dir * klStride + (size_t)b * n * n;
    buf[1][tid] = BIG_F;
    buf[2][tid] = BIG_F;
    __syncthreads();
    for (int k = 0; k < Kd; ++k) {
        float* rnew = buf[k % 3];
        const float* r1 = buf[(k + 2) % 3];   // diagonal k-1
        const float* r2 = buf[(k + 1) % 3];   // diagonal k-2
        if (tid < n) {
            float rl = r1[tid];                                    // (i, j-1)
            float ru = (tid == 0) ? BIG_F : r1[tid - 1];           // (i-1, j)
            float rd = (tid == 0) ? ((k == 0) ? 0.0f : BIG_F)
                                  : r2[tid - 1];                   // (i-1, j-1)
            float x0 = rd, x1 = ru + WARP_F, x2 = rl + WARP_F;
            float mn = fminf(x0, fminf(x1, x2));
            float s = __expf((mn - x0) / GAMMA_F)
                    + __expf((mn - x1) / GAMMA_F)
                    + __expf((mn - x2) / GAMMA_F);
            float smin = mn - GAMMA_F * __logf(s);
            int j = k - tid;
            rnew[tid] = (j >= 0 && j < n) ? (D[(size_t)tid * n + j] + smin) : BIG_F;
        }
        __syncthreads();
    }
    if (tid == 0) atomicAdd(&scr[4 + dir], buf[(Kd - 1) % 3][n - 1]);
}

__global__ void finalize_kernel(const float* __restrict__ scr,
                                const float* __restrict__ dur, const float* __restrict__ pitch,
                                float* out, int B_, int T_) {
    if (threadIdx.x != 0 || blockIdx.x != 0) return;
    float bt = (float)B_ * (float)T_;
    float g   = scr[0] / bt;                                   // GEN_ALPHA = 1
    float ge  = scr[1] / bt;                                   // GEN_E2E_ALPHA = 1
    float fm  = 4.0f * scr[2] / ((float)B_ * 128.0f * (float)T_);  // 2 * FEAT_ALPHA(2)
    float mel = 45.0f * scr[3] / ((float)B_ * 80.0f * (float)T_);  // MEL_ALPHA
    float du  = dur[0];
    float pi  = pitch[0];
    float kl1 = scr[4] / scr[6];
    float kl2 = scr[5] / scr[7];
    out[0] = g; out[1] = ge; out[2] = fm; out[3] = mel;
    out[4] = du; out[5] = pi; out[6] = kl1; out[7] = kl2;
    out[8] = g + ge + fm + mel + du + pi + kl1 + kl2;
}

static inline size_t align256(size_t x) { return (x + 255) & ~(size_t)255; }
static inline int nblk(long n) { long b = (n + 255) / 256; return (int)(b > 2048 ? 2048 : b); }

extern "C" void kernel_launch(void* const* d_in, const int* in_sizes, int n_in,
                              void* d_out, int out_size, void* d_ws, size_t ws_size,
                              hipStream_t stream) {
    const float* mel       = (const float*)d_in[0];
    const float* mel_hat   = (const float*)d_in[1];
    const float* sc        = (const float*)d_in[2];
    const float* sc_e2e    = (const float*)d_in[3];
    const float* fr        = (const float*)d_in[4];
    const float* ff        = (const float*)d_in[5];
    const float* dur       = (const float*)d_in[6];
    const float* pitch     = (const float*)d_in[7];
    const float* z_p       = (const float*)d_in[8];
    const float* m_p       = (const float*)d_in[9];
    const float* logs_p    = (const float*)d_in[10];
    const float* z_q       = (const float*)d_in[11];
    const float* m_q       = (const float*)d_in[12];
    const float* logs_q    = (const float*)d_in[13];
    const int*   p_mask    = (const int*)d_in[14];
    const int*   z_mask    = (const int*)d_in[15];

    const int B = 8;                                 // fixed in reference setup
    const int BT = in_sizes[14];                     // B*T
    const int T = BT / B;
    const int H = in_sizes[8] / BT;
    const int P16 = ((T + 15) / 16) * 16;
    const int K2 = 2 * H;
    const int Tp1 = T + 1;

    // ---- workspace layout ----
    char* w = (char*)d_ws;
    float* scr = (float*)w;
    size_t off = 256;
    unsigned short* Apack = (unsigned short*)(w + off); off += align256((size_t)B * P16 * K2 * 2);
    unsigned short* Bpack = (unsigned short*)(w + off); off += align256((size_t)B * P16 * K2 * 2);
    float* rowterm = (float*)(w + off); off += align256((size_t)B * P16 * 4);
    float* colterm = (float*)(w + off); off += align256((size_t)B * P16 * 4);
    float* klbuf   = (float*)(w + off);              // 2 directions, each B*Tp1*Tp1 f32
    size_t klStride = (size_t)B * Tp1 * Tp1;

    zero_scr_kernel<<<1, 32, 0, stream>>>(scr);

    // pointwise reductions
    long n_sc = in_sizes[2], n_fm = in_sizes[4], n_mel = in_sizes[0];
    reduce_sum_kernel<<<nblk(n_sc), 256, 0, stream>>>(sc, sc, n_sc, 0, scr + 0);
    reduce_sum_kernel<<<nblk(n_sc), 256, 0, stream>>>(sc_e2e, sc_e2e, n_sc, 0, scr + 1);
    reduce_sum_kernel<<<nblk(n_fm), 256, 0, stream>>>(fr, ff, n_fm, 1, scr + 2);
    reduce_sum_kernel<<<nblk(n_mel), 256, 0, stream>>>(mel, mel_hat, n_mel, 1, scr + 3);
    mask_sum_kernel<<<nblk(BT), 256, 0, stream>>>(p_mask, BT, scr + 6);
    mask_sum_kernel<<<nblk(BT), 256, 0, stream>>>(z_mask, BT, scr + 7);

    const int packBlocks = (B * P16 + 255) / 256;
    const int nt = P16 / 16;
    const int totalTiles = B * nt * nt;
    const int gemmBlocks = (totalTiles + 7) / 8;
    const int padBlocks = (B * Tp1 + 255) / 256;

    // direction 0: kl_loss_sdtw(z_p, logs_q, m_p, logs_p, p_mask, z_mask)
    pack_kernel<<<packBlocks, 256, 0, stream>>>(z_p, logs_q, m_p, logs_p,
                                                Apack, Bpack, rowterm, colterm, B, H, T, P16);
    kl_gemm_wmma_kernel<<<gemmBlocks, 256, 0, stream>>>(Apack, Bpack, rowterm, colterm,
                                                        p_mask, z_mask, klbuf, B, H, T, P16);
    pad_kernel<<<padBlocks, 256, 0, stream>>>(klbuf, p_mask, z_mask, B, T);

    // direction 1: kl_loss_sdtw(z_q, logs_p, m_q, logs_q, z_mask, p_mask)
    pack_kernel<<<packBlocks, 256, 0, stream>>>(z_q, logs_p, m_q, logs_q,
                                                Apack, Bpack, rowterm, colterm, B, H, T, P16);
    kl_gemm_wmma_kernel<<<gemmBlocks, 256, 0, stream>>>(Apack, Bpack, rowterm, colterm,
                                                        z_mask, p_mask, klbuf + klStride, B, H, T, P16);
    pad_kernel<<<padBlocks, 256, 0, stream>>>(klbuf + klStride, z_mask, p_mask, B, T);

    // both directions' soft-DTW wavefronts run concurrently (16 workgroups)
    dtw_kernel<<<2 * B, 512, 0, stream>>>(klbuf, B, T, klStride, scr);

    finalize_kernel<<<1, 32, 0, stream>>>(scr, dur, pitch, (float*)d_out, B, T);
}
